// LocalGlobalFusionStrong_19207093748297
// MI455X (gfx1250) — matrix operand order repaired and checked by reading, not compiled, not run on hardware
//
#include <hip/hip_runtime.h>
#include <stdint.h>
#include <stddef.h>

// ---------------------------------------------------------------------------
// PointNet++ MSG forward for MI455X (gfx1250, wave32, WMMA).
// - All GEMMs through v_wmma_f32_16x16x32_f16 (f32 accumulate).
// - SA levels: fused gather -> MLP(x3) -> maxpool, fully LDS-resident so the
//   O(100MB) grouped tensors never hit HBM.
// - Weights staged into LDS pre-swizzled fragment-major: B fragment = one
//   aligned v16h load (2x ds_load_b128).
// - Tile counts are template parameters so the multi-accumulator WMMA inner
//   loops are constant-bound, fully unrolled, branch-free straight-line code:
//   independent accumulators fill the F16 WMMA->WMMA hazard slots, and one
//   operand fragment is hoisted per wave (SA: B fixed; GEMM: A fixed).
// - No runtime integer division anywhere on device (shifts/masks only).
// ---------------------------------------------------------------------------

#define BATCH 4
#define NPTS  8192
#define CIN   6
#define SPTS  2048

typedef __attribute__((ext_vector_type(16))) _Float16 v16h;
typedef __attribute__((ext_vector_type(8)))  _Float16 v8h;
typedef __attribute__((ext_vector_type(8)))  float    v8f;

static __host__ __device__ inline int align32i(int x) { return (x + 31) & ~31; }

// ---- WMMA fragment loaders (contiguous-k order, same for A and B, wave32) -------------
// A: 16x32 f16, lane L holds row M=L&15; K = 16*(L>=16) + {0..15}.
// Two aligned 16B LDS loads per fragment.
static __device__ __forceinline__ v16h frag_a_lds(const _Float16* s, int stride,
                                                  int mBase, int kBase, int lane) {
  const _Float16* p = s + (size_t)(mBase + (lane & 15)) * stride + kBase + ((lane >> 4) << 4);
  v8h lo = *(const v8h*)p;
  v8h hi = *(const v8h*)(p + 8);
  return __builtin_shufflevector(lo, hi, 0, 1, 2, 3, 4, 5, 6, 7,
                                 8, 9, 10, 11, 12, 13, 14, 15);
}
// B fragments are staged swizzled: per (kChunk, nTile) a 32x16 block where
// lane L owns 16 contiguous halfs = col nBase+(L&15), K = kc + 16*(L>=16) + e.
// One aligned 32B LDS load per fragment.
static __device__ __forceinline__ v16h frag_b_swz(const _Float16* s, int fragIdx, int lane) {
  return *(const v16h*)(s + ((size_t)fragIdx << 9) + (lane << 4));
}

static __device__ __forceinline__ v8f v8f_zero() {
  v8f z = {0.f, 0.f, 0.f, 0.f, 0.f, 0.f, 0.f, 0.f};
  return z;
}

// ===========================================================================
// Farthest point sampling: one workgroup per batch, npoint sequential argmax
// rounds with an LDS tree reduction (first-max tie-break == jnp.argmax).
// ===========================================================================
__global__ void __launch_bounds__(256) fps_kernel(const float* __restrict__ xyz,
                                                  int n, int npoint, int* __restrict__ out) {
  const int b   = blockIdx.x;
  const int tid = threadIdx.x;
  const float* X = xyz + (size_t)b * n * 3;
  __shared__ float s_val[256];
  __shared__ int   s_idx[256];
  __shared__ float s_far[3];

  float dist[32];
#pragma unroll
  for (int i = 0; i < 32; ++i) dist[i] = 1e10f;
  const int npts = (n + 255) >> 8;   // points per thread (<=32)
  int far = 0;

  for (int it = 0; it < npoint; ++it) {
    if (tid == 0) {
      out[(size_t)b * npoint + it] = far;
      s_far[0] = X[far * 3 + 0]; s_far[1] = X[far * 3 + 1]; s_far[2] = X[far * 3 + 2];
    }
    __syncthreads();
    const float fx = s_far[0], fy = s_far[1], fz = s_far[2];
    float bv = -1.f; int bi = 0;
    for (int i = 0; i < npts; ++i) {
      const int p = tid + (i << 8);
      if (p < n) {
        const float dx = X[p * 3 + 0] - fx, dy = X[p * 3 + 1] - fy, dz = X[p * 3 + 2] - fz;
        float dd = dist[i];
        const float d = dx * dx + dy * dy + dz * dz;
        dd = d < dd ? d : dd;
        dist[i] = dd;
        if (dd > bv) { bv = dd; bi = p; }
      }
    }
    s_val[tid] = bv; s_idx[tid] = bi;
    __syncthreads();
    for (int s = 128; s > 0; s >>= 1) {
      if (tid < s) {
        const float ov = s_val[tid + s]; const int oi = s_idx[tid + s];
        if (ov > s_val[tid] || (ov == s_val[tid] && oi < s_idx[tid])) {
          s_val[tid] = ov; s_idx[tid] = oi;
        }
      }
      __syncthreads();
    }
    far = s_idx[0];
  }
}

__global__ void gather_xyz_kernel(const float* __restrict__ xyz, const int* __restrict__ fidx,
                                  int n, int npShift, int total, float* __restrict__ out) {
  const int gid = blockIdx.x * blockDim.x + threadIdx.x;
  if (gid >= total) return;
  const int b = gid >> npShift;
  const int p = fidx[gid];
  const float* s = xyz + ((size_t)b * n + p) * 3;
  out[(size_t)gid * 3 + 0] = s[0];
  out[(size_t)gid * 3 + 1] = s[1];
  out[(size_t)gid * 3 + 2] = s[2];
}

// Ball query: ns smallest in-radius indices in ascending index order.
__global__ void ball_query_kernel(const float* __restrict__ xyz, const float* __restrict__ nxyz,
                                  int n, int npShift, float r2, int ns, int total,
                                  int* __restrict__ out) {
  const int gid = blockIdx.x * blockDim.x + threadIdx.x;
  if (gid >= total) return;
  const int b = gid >> npShift;
  const float* X = xyz + (size_t)b * n * 3;
  const float cx = nxyz[(size_t)gid * 3 + 0];
  const float cy = nxyz[(size_t)gid * 3 + 1];
  const float cz = nxyz[(size_t)gid * 3 + 2];
  int* o = out + (size_t)gid * ns;
  int cnt = 0;
  for (int j = 0; j < n && cnt < ns; ++j) {
    const float dx = X[j * 3 + 0] - cx, dy = X[j * 3 + 1] - cy, dz = X[j * 3 + 2] - cz;
    if (dx * dx + dy * dy + dz * dz <= r2) o[cnt++] = j;
  }
  const int first = (cnt > 0) ? o[0] : 0;
  for (int k = cnt; k < ns; ++k) o[k] = first;
}

// ===========================================================================
// Fused SA kernel: gather (xyz-centered | feats) -> up to 3 dense+ReLU layers
// via WMMA in LDS -> maxpool over neighbors -> one pooled write to HBM.
// 128 group-rows per block (centers_per_block = 128/ns), 8 waves.
// Per-layer body templated on NTN = dout/16 so the WMMA loops are
// constant-bound and branch-free. Wave-fixed N-tile -> B fragment reused.
// ===========================================================================
struct SAArgs {
  const float* xyz; const float* feats; const float* nxyz; const int* idx;
  float* out; int out_stride; int col_off;
  int n, c, np, ns, nlayers;
  const float* W[3]; const float* Bb[3]; int dd[3];
};

template <int NTN>
static __device__ __forceinline__ void sa_mlp_layer(const _Float16* scur, _Float16* snxt,
                                                    _Float16* sw, float* sbias,
                                                    const float* __restrict__ W,
                                                    const float* __restrict__ Bv,
                                                    int din, int Kp,
                                                    int tid, int lane, int wave) {
  constexpr int ntNs = (NTN == 2) ? 1 : (NTN == 4) ? 2 : 3;
  constexpr int DOUT = NTN << 4;
  constexpr int DS   = ntNs + 4;
  constexpr int mtS  = 8 >> ntNs;

  // stage weights swizzled fragment-major:
  // frag (kc/32, nt): lane dl = 16*(kin>=16) + (j&15), elem e = kin&15.
  for (int t = tid; t < Kp * DOUT; t += 256) {
    const int k = t >> DS, j = t & (DOUT - 1);
    const float v = (k < din) ? W[(size_t)k * DOUT + j] : 0.f;
    const int kcIdx = k >> 5, kin = k & 31;
    const int dl = ((kin >> 4) << 4) | (j & 15);
    const int fragIdx = (kcIdx << ntNs) | (j >> 4);
    sw[((size_t)fragIdx << 9) + (dl << 4) + (kin & 15)] = (_Float16)v;
  }
  for (int t = tid; t < DOUT; t += 256) sbias[t] = Bv[t];
  __syncthreads();

  const int myNt = wave & (NTN - 1);
  const int mtB  = wave >> ntNs;
  v8f acc[NTN];
#pragma unroll
  for (int i = 0; i < NTN; ++i) acc[i] = v8f_zero();

  for (int kc = 0; kc < Kp; kc += 32) {
    const v16h bm = frag_b_swz(sw, ((kc >> 5) << ntNs) | myNt, lane);  // reused
#pragma unroll
    for (int t = 0; t < NTN; ++t) {
      v16h a = frag_a_lds(scur, Kp, (mtB + mtS * t) << 4, kc, lane);
      acc[t] = __builtin_amdgcn_wmma_f32_16x16x32_f16(false, a, false, bm,
                                                      (short)0, acc[t], false, false);
    }
  }
  // C layout: vgpr i -> row mBase + 8*(lane>=16) + i, col nBase + (lane&15)
  const int col = (myNt << 4) + (lane & 15);
  const float bv = sbias[col];
#pragma unroll
  for (int t = 0; t < NTN; ++t) {
    const int rb = ((mtB + mtS * t) << 4) + ((lane >> 4) << 3);
#pragma unroll
    for (int i = 0; i < 8; ++i) {
      float v = acc[t][i] + bv;
      v = v > 0.f ? v : 0.f;                      // ReLU (all SA layers)
      snxt[(size_t)(rb + i) * DOUT + col] = (_Float16)v;
    }
  }
  __syncthreads();
}

__global__ void __launch_bounds__(256) sa_fused_kernel(SAArgs A, int Kmax, int dmax) {
  extern __shared__ char smem[];
  _Float16* sb0   = (_Float16*)smem;
  _Float16* sb1   = sb0 + (size_t)128 * Kmax;
  _Float16* sw    = sb1 + (size_t)128 * Kmax;          // swizzled weights
  float*    sbias = (float*)(sw + (size_t)Kmax * dmax);

  const int tid = threadIdx.x, lane = tid & 31, wave = tid >> 5;
  const int b = blockIdx.y;
  const int lns = 31 - __builtin_clz((unsigned)A.ns);  // ns is pow2 (4..32)
  const int cpb = 128 >> lns;
  const int cbase = blockIdx.x * cpb;

  // ---- gather grouped rows into LDS (f16), zero-padded to K0 ----
  const int din0 = 3 + A.c;
  const int K0 = align32i(din0);
  for (int r = wave; r < 128; r += 8) {
    const int ci = r >> lns, s = r & (A.ns - 1);
    const int gc = cbase + ci;
    const bool ok = (gc < A.np);
    const int p = ok ? A.idx[((size_t)b * A.np + gc) * A.ns + s] : 0;
    const float* xr = A.xyz  + ((size_t)b * A.n + p) * 3;
    const float* cr = A.nxyz + ((size_t)b * A.np + (ok ? gc : 0)) * 3;
    const float* fr = A.feats + ((size_t)b * A.n + p) * A.c;
    for (int k = lane; k < K0; k += 32) {
      float v = 0.f;
      if (ok && k < din0) v = (k < 3) ? (xr[k] - cr[k]) : fr[k - 3];
      sb0[(size_t)r * K0 + k] = (_Float16)v;
    }
  }
  __syncthreads();

  // ---- MLP layers, LDS-resident ping-pong; block-uniform switch on dout ----
  _Float16* scur = sb0; _Float16* snxt = sb1;
  int din = din0;
  for (int l = 0; l < A.nlayers; ++l) {
    const int dout = A.dd[l];                     // 32 / 64 / 128
    const int Kp = align32i(din);
    if (tid == 0 && l + 1 < A.nlayers) __builtin_prefetch(A.W[l + 1], 0, 0);
    switch (dout) {
      case 32:
        sa_mlp_layer<2>(scur, snxt, sw, sbias, A.W[l], A.Bb[l], din, Kp, tid, lane, wave);
        break;
      case 64:
        sa_mlp_layer<4>(scur, snxt, sw, sbias, A.W[l], A.Bb[l], din, Kp, tid, lane, wave);
        break;
      default:
        sa_mlp_layer<8>(scur, snxt, sw, sbias, A.W[l], A.Bb[l], din, Kp, tid, lane, wave);
        break;
    }
    _Float16* tmpp = scur; scur = snxt; snxt = tmpp;
    din = dout;
  }

  // ---- maxpool over ns neighbors, write pooled result ----
  const int df  = din;                             // pow2: 64/128
  const int dfs = 31 - __builtin_clz((unsigned)df);
  for (int t = tid; t < (cpb << dfs); t += 256) {
    const int ci = t >> dfs, j = t & (df - 1);
    const int gc = cbase + ci;
    if (gc < A.np) {
      float m = -3.4e38f;
      for (int s = 0; s < A.ns; ++s) {
        const float v = (float)scur[(size_t)(((ci << lns) + s)) * df + j];
        m = v > m ? v : m;
      }
      A.out[((size_t)b * A.np + gc) * A.out_stride + A.col_off + j] = m;
    }
  }
}

// ===========================================================================
// Generic GEMM: rows x (c0|c1 concat) @ W(din,dout) + bias, optional ReLU.
// Templated on TPW = dout/32 (tiles per wave). 64-row tiles, K staged in
// 32-deep swizzled LDS chunks. nt-major tile map: mt = wave & 3 fixed per
// wave (A fragment hoisted), nt = (wave>>2) + 2*t -> independent accs.
// ===========================================================================
struct GemmArgs {
  const float* in0; int c0; const float* in1; int c1;
  const float* W; const float* bias;
  float* out; int nrows; int relu;
};

template <int TPW>
__global__ void __launch_bounds__(256) gemm_kernel(GemmArgs G, int Kp) {
  constexpr int DOUT = TPW << 5;                  // 64 / 128 / 256
  constexpr int DS   = (TPW == 2) ? 6 : (TPW == 4) ? 7 : 8;
  extern __shared__ char smem[];
  _Float16* sa = (_Float16*)smem;                 // 64 x Kp
  _Float16* sw = sa + (size_t)64 * Kp;            // one swizzled 32 x DOUT chunk
  float* sbias = (float*)(sw + (size_t)32 * DOUT);

  const int tid = threadIdx.x, lane = tid & 31, wave = tid >> 5;
  const int r0 = blockIdx.x * 64;
  const int din = G.c0 + G.c1;

  for (int r = wave; r < 64; r += 8) {
    const int gr = r0 + r;
    const bool ok = gr < G.nrows;
    for (int k = lane; k < Kp; k += 32) {
      float v = 0.f;
      if (ok) {
        if (k < G.c0)     v = G.in0[(size_t)gr * G.c0 + k];
        else if (k < din) v = G.in1[(size_t)gr * G.c1 + (k - G.c0)];
      }
      sa[(size_t)r * Kp + k] = (_Float16)v;
    }
  }
  for (int t = tid; t < DOUT; t += 256) sbias[t] = G.bias[t];

  const int myMt = wave & 3;                      // fixed M-tile per wave
  const int ntB  = wave >> 2;                     // 0 or 1
  v8f acc[TPW];
#pragma unroll
  for (int i = 0; i < TPW; ++i) acc[i] = v8f_zero();

  for (int kc = 0; kc < Kp; kc += 32) {
    __syncthreads();                              // protect sw reuse / first sa use
    if (lane == 0 && kc + 32 < Kp)                // prefetch next W chunk
      __builtin_prefetch(G.W + (size_t)(kc + 32) * DOUT, 0, 0);
    for (int t = tid; t < (DOUT << 5); t += 256) {
      const int k = t >> DS, j = t & (DOUT - 1);
      const int gk = kc + k;
      const float v = (gk < din) ? G.W[(size_t)gk * DOUT + j] : 0.f;
      const int dl = ((k >> 4) << 4) | (j & 15);
      sw[((size_t)(j >> 4) << 9) + (dl << 4) + (k & 15)] = (_Float16)v;
    }
    __syncthreads();
    const v16h a = frag_a_lds(sa, Kp, myMt << 4, kc, lane);   // reused across nt
#pragma unroll
    for (int t = 0; t < TPW; ++t) {
      v16h bm = frag_b_swz(sw, ntB + 2 * t, lane);
      acc[t] = __builtin_amdgcn_wmma_f32_16x16x32_f16(false, a, false, bm,
                                                      (short)0, acc[t], false, false);
    }
  }
  const int rb = (myMt << 4) + ((lane >> 4) << 3);
#pragma unroll
  for (int t = 0; t < TPW; ++t) {
    const int col = ((ntB + 2 * t) << 4) + (lane & 15);
    const float bv = sbias[col];
#pragma unroll
    for (int i = 0; i < 8; ++i) {
      const int gr = r0 + rb + i;
      if (gr < G.nrows) {
        float v = acc[t][i] + bv;
        if (G.relu) v = v > 0.f ? v : 0.f;
        G.out[(size_t)gr * DOUT + col] = v;
      }
    }
  }
}

// ===========================================================================
// Feature propagation: 3-NN + inverse-distance weights, then interpolation.
// ===========================================================================
__global__ void knn3_kernel(const float* __restrict__ xyz1, const float* __restrict__ xyz2,
                            int n1Shift, int n2, int total, int* __restrict__ oidx,
                            float* __restrict__ ow) {
  const int gid = blockIdx.x * blockDim.x + threadIdx.x;
  if (gid >= total) return;
  const int b = gid >> n1Shift;
  const float* X2 = xyz2 + (size_t)b * n2 * 3;
  const float px = xyz1[(size_t)gid * 3 + 0];
  const float py = xyz1[(size_t)gid * 3 + 1];
  const float pz = xyz1[(size_t)gid * 3 + 2];
  float d0 = 1e30f, d1 = 1e30f, d2 = 1e30f;
  int i0 = 0, i1 = 0, i2 = 0;
  for (int j = 0; j < n2; ++j) {
    const float dx = X2[j * 3 + 0] - px, dy = X2[j * 3 + 1] - py, dz = X2[j * 3 + 2] - pz;
    const float d = dx * dx + dy * dy + dz * dz;
    if (d < d0)      { d2 = d1; i2 = i1; d1 = d0; i1 = i0; d0 = d; i0 = j; }
    else if (d < d1) { d2 = d1; i2 = i1; d1 = d; i1 = j; }
    else if (d < d2) { d2 = d; i2 = j; }
  }
  float w0 = 1.f / (fmaxf(d0, 0.f) + 1e-8f);
  float w1 = 1.f / (fmaxf(d1, 0.f) + 1e-8f);
  float w2 = 1.f / (fmaxf(d2, 0.f) + 1e-8f);
  const float s = w0 + w1 + w2;
  ow[(size_t)gid * 3 + 0] = w0 / s;
  ow[(size_t)gid * 3 + 1] = w1 / s;
  ow[(size_t)gid * 3 + 2] = w2 / s;
  oidx[(size_t)gid * 3 + 0] = i0;
  oidx[(size_t)gid * 3 + 1] = i1;
  oidx[(size_t)gid * 3 + 2] = i2;
}

__global__ void interp_kernel(const float* __restrict__ f2, const int* __restrict__ kidx,
                              const float* __restrict__ kw, int c2Shift, int n1Shift,
                              int n2, int total, float* __restrict__ out) {
  const int gid = blockIdx.x * blockDim.x + threadIdx.x;
  if (gid >= total) return;
  const int c2 = 1 << c2Shift;
  const int j = gid & (c2 - 1);
  const int r = gid >> c2Shift;
  const int b = r >> n1Shift;
  const float* F = f2 + (((size_t)b * n2) << c2Shift);
  out[gid] = kw[(size_t)r * 3 + 0] * F[((size_t)kidx[(size_t)r * 3 + 0] << c2Shift) + j]
           + kw[(size_t)r * 3 + 1] * F[((size_t)kidx[(size_t)r * 3 + 1] << c2Shift) + j]
           + kw[(size_t)r * 3 + 2] * F[((size_t)kidx[(size_t)r * 3 + 2] << c2Shift) + j];
}

// Global stats over l4f (np x 256 per batch): per-channel mean/std(ddof=1)/
// max/min/range, averaged over channels, broadcast to np x 5.
__global__ void __launch_bounds__(256) stats_kernel(const float* __restrict__ f, int np, int c,
                                                    float* __restrict__ out) {
  const int b = blockIdx.x;
  const int ch = threadIdx.x;
  __shared__ float sred[5][256];
  __shared__ float sv[5];
  float sum = 0.f, sq = 0.f, mx = -3.4e38f, mn = 3.4e38f;
  const float* F = f + (size_t)b * np * c + ch;
  for (int i = 0; i < np; ++i) {
    const float x = F[(size_t)i * c];
    sum += x; sq += x * x;
    mx = fmaxf(mx, x); mn = fminf(mn, x);
  }
  const float mean = sum / (float)np;
  float var = (sq - (float)np * mean * mean) / (float)(np - 1);
  var = fmaxf(var, 0.f);
  sred[0][ch] = mean; sred[1][ch] = sqrtf(var); sred[2][ch] = mx;
  sred[3][ch] = mn;   sred[4][ch] = mx - mn;
  __syncthreads();
  for (int s = 128; s > 0; s >>= 1) {
    if (ch < s) {
#pragma unroll
      for (int k = 0; k < 5; ++k) sred[k][ch] += sred[k][ch + s];
    }
    __syncthreads();
  }
  if (ch < 5) sv[ch] = sred[ch][0] / (float)c;
  __syncthreads();
  for (int i = ch; i < np * 5; i += 256) out[(size_t)b * np * 5 + i] = sv[i % 5];
}

__global__ void transpose_kernel(const float* __restrict__ in, float* __restrict__ out,
                                 int nShift, int cShift, int total) {
  const int gid = blockIdx.x * blockDim.x + threadIdx.x;
  if (gid >= total) return;
  const int i = gid & ((1 << nShift) - 1);
  const int t = gid >> nShift;
  const int ch = t & ((1 << cShift) - 1);
  const int b = t >> cShift;
  out[gid] = in[((((size_t)b << nShift) + i) << cShift) + ch];
}

// ===========================================================================
// Host orchestration
// ===========================================================================
static inline int imax2(int a, int b) { return a > b ? a : b; }

static void launch_sa(hipStream_t st, const float* xyz, const float* feats,
                      const float* nxyz, const int* idx, float* out, int ostride,
                      int coloff, int n, int c, int np, int ns, int nl,
                      const float* w0, const float* b0, int d0,
                      const float* w1, const float* b1, int d1,
                      const float* w2, const float* b2, int d2) {
  SAArgs A;
  A.xyz = xyz; A.feats = feats; A.nxyz = nxyz; A.idx = idx;
  A.out = out; A.out_stride = ostride; A.col_off = coloff;
  A.n = n; A.c = c; A.np = np; A.ns = ns; A.nlayers = nl;
  A.W[0] = w0; A.Bb[0] = b0; A.dd[0] = d0;
  A.W[1] = w1; A.Bb[1] = b1; A.dd[1] = d1;
  A.W[2] = w2; A.Bb[2] = b2; A.dd[2] = d2;
  int Kmax = align32i(3 + c);
  Kmax = imax2(Kmax, align32i(d0));
  Kmax = imax2(Kmax, align32i(d1));
  Kmax = imax2(Kmax, align32i(d2));
  int dmax = imax2(d0, imax2(d1, d2));
  size_t shmem = ((size_t)2 * 128 * Kmax + (size_t)Kmax * dmax) * sizeof(_Float16)
               + (size_t)dmax * sizeof(float);
  const int cpb = 128 / ns;
  dim3 grid((np + cpb - 1) / cpb, BATCH);
  sa_fused_kernel<<<grid, 256, shmem, st>>>(A, Kmax, dmax);
}

static void launch_gemm(hipStream_t st, const float* in0, int c0, const float* in1,
                        int c1, const float* W, const float* bias, float* out,
                        int dout, int nrows, int relu) {
  GemmArgs G;
  G.in0 = in0; G.c0 = c0; G.in1 = in1; G.c1 = c1;
  G.W = W; G.bias = bias; G.out = out; G.nrows = nrows; G.relu = relu;
  const int Kp = align32i(c0 + c1);
  size_t shmem = ((size_t)64 * Kp + (size_t)32 * dout) * sizeof(_Float16)
               + (size_t)dout * sizeof(float);
  dim3 grid((nrows + 63) / 64);
  switch (dout) {
    case 64:  gemm_kernel<2><<<grid, 256, shmem, st>>>(G, Kp); break;
    case 128: gemm_kernel<4><<<grid, 256, shmem, st>>>(G, Kp); break;
    default:  gemm_kernel<8><<<grid, 256, shmem, st>>>(G, Kp); break;
  }
}

extern "C" void kernel_launch(void* const* d_in, const int* in_sizes, int n_in,
                              void* d_out, int out_size, void* d_ws, size_t ws_size,
                              hipStream_t stream) {
  (void)in_sizes; (void)n_in; (void)out_size; (void)ws_size;

  const float* pc    = (const float*)d_in[0];   // B x N x 3
  const float* feat0 = (const float*)d_in[1];   // B x N x 6
#define PW(i) ((const float*)d_in[(i)])
  // Param flat order (setup_inputs insertion order, depth-first):
  // sa1[0]: 2..7  sa1[1]: 8..13  lin1: 14,15
  // sa2[0]: 16..19 sa2[1]: 20..23 lin2: 24,25
  // sa3[0]: 26..29 sa3[1]: 30..33 lin3: 34,35
  // sa4[0]: 36..39 sa4[1]: 40..43 lin4: 44,45
  // fusion: 46,47  fp4: 48,49  fp3: 50,51  fp2: 52,53  fp1: 54,55

  // ---- workspace bump allocator ----
  char* wp = (char*)d_ws;
  auto af = [&](size_t nfl) { float* r = (float*)wp; wp += ((nfl * 4 + 255) & ~(size_t)255); return r; };
  auto ai = [&](size_t nin) { int*   r = (int*)wp;   wp += ((nin * 4 + 255) & ~(size_t)255); return r; };

  int*   fpsidx  = ai((size_t)BATCH * SPTS);
  int*   ballidx = ai((size_t)BATCH * SPTS * 32);
  int*   kidx    = ai((size_t)BATCH * NPTS * 3);
  float* kw      = af((size_t)BATCH * NPTS * 3);
  float* l1x     = af((size_t)BATCH * SPTS * 3);
  float* l2x     = af((size_t)BATCH * SPTS * 3);
  float* l3x     = af((size_t)BATCH * SPTS * 3);
  float* l1m     = af((size_t)BATCH * SPTS * 128);
  float* l1f     = af((size_t)BATCH * SPTS * 64);
  float* l2m     = af((size_t)BATCH * SPTS * 192);
  float* l2f     = af((size_t)BATCH * SPTS * 128);
  float* l3m     = af((size_t)BATCH * SPTS * 256);
  float* l3f     = af((size_t)BATCH * SPTS * 128);
  float* l4m     = af((size_t)BATCH * 1024 * 256);
  float* l4f     = af((size_t)BATCH * 1024 * 256);
  float* stats   = af((size_t)BATCH * 1024 * 5);
  float* l4fu    = af((size_t)BATCH * 1024 * 256);
  float* interp  = af((size_t)BATCH * NPTS * 128);   // max of all fp interp uses
  float* l3f2    = af((size_t)BATCH * SPTS * 256);
  float* l2f2    = af((size_t)BATCH * SPTS * 128);
  float* l1f2    = af((size_t)BATCH * SPTS * 128);
  float* l0f     = af((size_t)BATCH * NPTS * 128);

  float* outF = (float*)d_out;
  float* l4x  = outF;                       // B x 1024 x 3 written in place
  float* l0fT = outF + (size_t)BATCH * 1024 * 3;

  const int TB = 256;
  auto gblk = [&](int total) { return (total + TB - 1) / TB; };

  // ---------------- Level 1: n=8192, c=6, np=2048 ----------------
  fps_kernel<<<BATCH, 256, 0, stream>>>(pc, NPTS, SPTS, fpsidx);
  gather_xyz_kernel<<<gblk(BATCH * SPTS), TB, 0, stream>>>(pc, fpsidx, NPTS, 11, BATCH * SPTS, l1x);
  ball_query_kernel<<<gblk(BATCH * SPTS), TB, 0, stream>>>(pc, l1x, NPTS, 11, 4.0f, 4, BATCH * SPTS, ballidx);
  launch_sa(stream, pc, feat0, l1x, ballidx, l1m, 128, 0, NPTS, CIN, SPTS, 4, 3,
            PW(2), PW(3), 32, PW(4), PW(5), 32, PW(6), PW(7), 64);
  ball_query_kernel<<<gblk(BATCH * SPTS), TB, 0, stream>>>(pc, l1x, NPTS, 11, 16.0f, 8, BATCH * SPTS, ballidx);
  launch_sa(stream, pc, feat0, l1x, ballidx, l1m, 128, 64, NPTS, CIN, SPTS, 8, 3,
            PW(8), PW(9), 32, PW(10), PW(11), 32, PW(12), PW(13), 64);
  launch_gemm(stream, l1m, 128, nullptr, 0, PW(14), PW(15), l1f, 64, BATCH * SPTS, 0);

  // ---------------- Level 2: n=2048, c=64, np=2048 ----------------
  fps_kernel<<<BATCH, 256, 0, stream>>>(l1x, SPTS, SPTS, fpsidx);
  gather_xyz_kernel<<<gblk(BATCH * SPTS), TB, 0, stream>>>(l1x, fpsidx, SPTS, 11, BATCH * SPTS, l2x);
  ball_query_kernel<<<gblk(BATCH * SPTS), TB, 0, stream>>>(l1x, l2x, SPTS, 11, 16.0f, 8, BATCH * SPTS, ballidx);
  launch_sa(stream, l1x, l1f, l2x, ballidx, l2m, 192, 0, SPTS, 64, SPTS, 8, 2,
            PW(16), PW(17), 64, PW(18), PW(19), 64, nullptr, nullptr, 0);
  ball_query_kernel<<<gblk(BATCH * SPTS), TB, 0, stream>>>(l1x, l2x, SPTS, 11, 64.0f, 16, BATCH * SPTS, ballidx);
  launch_sa(stream, l1x, l1f, l2x, ballidx, l2m, 192, 64, SPTS, 64, SPTS, 16, 2,
            PW(20), PW(21), 64, PW(22), PW(23), 128, nullptr, nullptr, 0);
  launch_gemm(stream, l2m, 192, nullptr, 0, PW(24), PW(25), l2f, 128, BATCH * SPTS, 0);

  // ---------------- Level 3: n=2048, c=128, np=2048 ----------------
  fps_kernel<<<BATCH, 256, 0, stream>>>(l2x, SPTS, SPTS, fpsidx);
  gather_xyz_kernel<<<gblk(BATCH * SPTS), TB, 0, stream>>>(l2x, fpsidx, SPTS, 11, BATCH * SPTS, l3x);
  ball_query_kernel<<<gblk(BATCH * SPTS), TB, 0, stream>>>(l2x, l3x, SPTS, 11, 64.0f, 16, BATCH * SPTS, ballidx);
  launch_sa(stream, l2x, l2f, l3x, ballidx, l3m, 256, 0, SPTS, 128, SPTS, 16, 2,
            PW(26), PW(27), 128, PW(28), PW(29), 128, nullptr, nullptr, 0);
  ball_query_kernel<<<gblk(BATCH * SPTS), TB, 0, stream>>>(l2x, l3x, SPTS, 11, 256.0f, 32, BATCH * SPTS, ballidx);
  launch_sa(stream, l2x, l2f, l3x, ballidx, l3m, 256, 128, SPTS, 128, SPTS, 32, 2,
            PW(30), PW(31), 128, PW(32), PW(33), 128, nullptr, nullptr, 0);
  launch_gemm(stream, l3m, 256, nullptr, 0, PW(34), PW(35), l3f, 128, BATCH * SPTS, 0);

  // ---------------- Level 4: n=2048, c=128, np=1024 ----------------
  fps_kernel<<<BATCH, 256, 0, stream>>>(l3x, SPTS, 1024, fpsidx);
  gather_xyz_kernel<<<gblk(BATCH * 1024), TB, 0, stream>>>(l3x, fpsidx, SPTS, 10, BATCH * 1024, l4x);
  ball_query_kernel<<<gblk(BATCH * 1024), TB, 0, stream>>>(l3x, l4x, SPTS, 10, 256.0f, 16, BATCH * 1024, ballidx);
  launch_sa(stream, l3x, l3f, l4x, ballidx, l4m, 256, 0, SPTS, 128, 1024, 16, 2,
            PW(36), PW(37), 128, PW(38), PW(39), 128, nullptr, nullptr, 0);
  ball_query_kernel<<<gblk(BATCH * 1024), TB, 0, stream>>>(l3x, l4x, SPTS, 10, 1024.0f, 32, BATCH * 1024, ballidx);
  launch_sa(stream, l3x, l3f, l4x, ballidx, l4m, 256, 128, SPTS, 128, 1024, 32, 2,
            PW(40), PW(41), 128, PW(42), PW(43), 128, nullptr, nullptr, 0);
  launch_gemm(stream, l4m, 256, nullptr, 0, PW(44), PW(45), l4f, 256, BATCH * 1024, 0);

  // ---------------- global stats + fusion ----------------
  stats_kernel<<<BATCH, 256, 0, stream>>>(l4f, 1024, 256, stats);
  launch_gemm(stream, l4f, 256, stats, 5, PW(46), PW(47), l4fu, 256, BATCH * 1024, 0);

  // ---------------- feature propagation ----------------
  // fp4: l3 (2048) <- l4 (1024), concat(l3f[128], interp(l4fu)[256]) -> 256
  knn3_kernel<<<gblk(BATCH * SPTS), TB, 0, stream>>>(l3x, l4x, 11, 1024, BATCH * SPTS, kidx, kw);
  interp_kernel<<<gblk(BATCH * SPTS * 256), TB, 0, stream>>>(l4fu, kidx, kw, 8, 11, 1024,
                                                             BATCH * SPTS * 256, interp);
  launch_gemm(stream, l3f, 128, interp, 256, PW(48), PW(49), l3f2, 256, BATCH * SPTS, 1);

  // fp3: l2 <- l3, concat(l2f[128], interp(l3f2)[256]) -> 128
  knn3_kernel<<<gblk(BATCH * SPTS), TB, 0, stream>>>(l2x, l3x, 11, SPTS, BATCH * SPTS, kidx, kw);
  interp_kernel<<<gblk(BATCH * SPTS * 256), TB, 0, stream>>>(l3f2, kidx, kw, 8, 11, SPTS,
                                                             BATCH * SPTS * 256, interp);
  launch_gemm(stream, l2f, 128, interp, 256, PW(50), PW(51), l2f2, 128, BATCH * SPTS, 1);

  // fp2: l1 <- l2, concat(l1f[64], interp(l2f2)[128]) -> 128
  knn3_kernel<<<gblk(BATCH * SPTS), TB, 0, stream>>>(l1x, l2x, 11, SPTS, BATCH * SPTS, kidx, kw);
  interp_kernel<<<gblk(BATCH * SPTS * 128), TB, 0, stream>>>(l2f2, kidx, kw, 7, 11, SPTS,
                                                             BATCH * SPTS * 128, interp);
  launch_gemm(stream, l1f, 64, interp, 128, PW(52), PW(53), l1f2, 128, BATCH * SPTS, 1);

  // fp1: l0 (8192) <- l1, interp(l1f2)[128] -> 128
  knn3_kernel<<<gblk(BATCH * NPTS), TB, 0, stream>>>(pc, l1x, 13, SPTS, BATCH * NPTS, kidx, kw);
  interp_kernel<<<gblk(BATCH * NPTS * 128), TB, 0, stream>>>(l1f2, kidx, kw, 7, 13, SPTS,
                                                             BATCH * NPTS * 128, interp);
  launch_gemm(stream, interp, 128, nullptr, 0, PW(54), PW(55), l0f, 128, BATCH * NPTS, 1);

  // ---------------- output: (l4x already in d_out) + transpose(l0f) ----------
  transpose_kernel<<<gblk(BATCH * 128 * NPTS), TB, 0, stream>>>(l0f, l0fT, 13, 7,
                                                                BATCH * 128 * NPTS);
#undef PW
}
